// HGCNConv_64510408786254
// MI455X (gfx1250) — matrix operand order repaired and checked by reading, not compiled, not run on hardware
//
#include <hip/hip_runtime.h>
#include <stdint.h>

typedef __attribute__((ext_vector_type(16))) _Float16 v16h;
typedef __attribute__((ext_vector_type(8)))  _Float16 v8h;
typedef __attribute__((ext_vector_type(8)))  float    v8f;

#define DIM 128
#define MINN 1e-15f
#define MAXN (1.0f - 4e-3f)

// ---------------------------------------------------------------- helpers
__device__ __forceinline__ float wave_sum(float v) {
    #pragma unroll
    for (int off = 16; off > 0; off >>= 1) v += __shfl_xor(v, off, 32);
    return v;
}

__device__ __forceinline__ float artanh_f(float x) {
    x = fminf(fmaxf(x, -1.0f + 1e-7f), 1.0f - 1e-7f);
    return 0.5f * (log1pf(x) - log1pf(-x));
}

__device__ __forceinline__ float dot4(float4 a, float4 b) {
    return a.x * b.x + a.y * b.y + a.z * b.z + a.w * b.w;
}
__device__ __forceinline__ float4 scale4(float4 a, float s) {
    return make_float4(a.x * s, a.y * s, a.z * s, a.w * s);
}
__device__ __forceinline__ float4 fma4(float4 a, float sa, float4 b, float sb) {
    return make_float4(a.x * sa + b.x * sb, a.y * sa + b.y * sb,
                       a.z * sa + b.z * sb, a.w * sa + b.w * sb);
}

// ---------------------------------------------------------------- K0: W f32 -> f16
__global__ __launch_bounds__(256) void k_convert_w(const float* __restrict__ w,
                                                   _Float16* __restrict__ wh, int n) {
    int i = blockIdx.x * 256 + threadIdx.x;
    if (i < n) wh[i] = (_Float16)w[i];
}

// ---------------------------------------------------------------- K1: WMMA GEMM  mx = x @ W^T
// Block = 256 thr = 8 waves; each block computes a 128x128 output tile.
// B[k][n] = W[n][k]; sW stored row-major like W (sW[n*128+k]) so B-fragment reads are contiguous.
// W tile staged via CDNA5 async copy (GLOBAL_LOAD_ASYNC_TO_LDS_B128, ASYNCcnt-tracked).
__global__ __launch_bounds__(256) void k_gemm_wmma(const float* __restrict__ x,
                                                   const _Float16* __restrict__ wh,
                                                   float* __restrict__ mx, int N) {
    __shared__ _Float16 sX[128 * 128]; // [row][k]
    __shared__ _Float16 sW[128 * 128]; // [n][k]
    const int tid = threadIdx.x;
    const int row0 = blockIdx.x * 128;

    // stage W: 32KB = 2048 x 16B async transfers, no VGPR round-trip.
    // each lane of the async op moves 16B: LDS[vdst_lane] <- MEM[vaddr_lane]
    #pragma unroll
    for (int i = tid; i < 128 * 128 / 8; i += 256) {
        unsigned lds_addr = (unsigned)(uintptr_t)(&sW[i * 8]);
        uint64_t gaddr    = (uint64_t)(uintptr_t)(wh + i * 8);
        asm volatile("global_load_async_to_lds_b128 %0, %1, off"
                     :: "v"(lds_addr), "v"(gaddr) : "memory");
    }

    // stage x tile, f32 -> f16 conversion in flight, zero-pad rows >= N
    for (int i = tid; i < 128 * 128 / 4; i += 256) {
        int r = i >> 5;          // 32 float4 per row
        int c4 = (i & 31);
        int gr = row0 + r;
        float4 v = (gr < N) ? ((const float4*)x)[gr * 32 + c4]
                            : make_float4(0.f, 0.f, 0.f, 0.f);
        _Float16* d = &sX[r * 128 + c4 * 4];
        d[0] = (_Float16)v.x; d[1] = (_Float16)v.y;
        d[2] = (_Float16)v.z; d[3] = (_Float16)v.w;
    }
    asm volatile("s_wait_asynccnt 0x0" ::: "memory");
    __syncthreads();

    const int wave = tid >> 5;
    const int lane = tid & 31;
    const int m0 = wave * 16;            // local row base of this wave's 16x128 strip
    const int lh = lane & 15;
    const int kA = (lane < 16) ? 0 : 8;  // A frag K sub-offset (16-bit A 16x32 layout)
    const int kB = (lane < 16) ? 0 : 8;  // B frag K sub-offset (16-bit B 32x16 layout)

    v8f acc[8];
    #pragma unroll
    for (int nt = 0; nt < 8; ++nt)
        #pragma unroll
        for (int r = 0; r < 8; ++r) acc[nt][r] = 0.0f;

    #pragma unroll
    for (int ks = 0; ks < 4; ++ks) {
        const int kbase = ks * 32;
        // A fragment: row m0+lh, halves [kA..kA+7] and [kA+16..kA+23] (16B aligned)
        const _Float16* arow = &sX[(m0 + lh) * 128 + kbase];
        v8h a0 = *(const v8h*)(arow + kA);
        v8h a1 = *(const v8h*)(arow + kA + 16);
        v16h a = __builtin_shufflevector(a0, a1, 0, 1, 2, 3, 4, 5, 6, 7,
                                         8, 9, 10, 11, 12, 13, 14, 15);
        // hoist all 8 B fragments so the 8 WMMAs can issue back-to-back
        v16h b[8];
        #pragma unroll
        for (int nt = 0; nt < 8; ++nt) {
            const _Float16* bcol = &sW[(nt * 16 + lh) * 128 + kbase + kB];
            v8h b0 = *(const v8h*)(bcol);
            v8h b1 = *(const v8h*)(bcol + 16);
            b[nt] = __builtin_shufflevector(b0, b1, 0, 1, 2, 3, 4, 5, 6, 7,
                                            8, 9, 10, 11, 12, 13, 14, 15);
        }
        #pragma unroll
        for (int nt = 0; nt < 8; ++nt)
            acc[nt] = __builtin_amdgcn_wmma_f32_16x16x32_f16(
                false, a, false, b[nt], (short)0, acc[nt], false, false);
    }

    // D layout: VGPR r -> (M=r, N=lane) lanes 0-15 ; (M=r+8, N=lane-16) lanes 16-31
    const int nl = lane & 15;
    const int mh = (lane >> 4) * 8;
    #pragma unroll
    for (int nt = 0; nt < 8; ++nt)
        #pragma unroll
        for (int r = 0; r < 8; ++r) {
            int gr = row0 + m0 + mh + r;
            if (gr < N) mx[gr * 128 + nt * 16 + nl] = acc[nt][r];
        }
}

// ---------------------------------------------------------------- K2: HypLinear tail + logmap0
// one wave per row; buf holds mx in, xt out (in place). c = 1.
__global__ __launch_bounds__(256) void k_linear_tail(const float* __restrict__ x,
                                                     const float* __restrict__ bias,
                                                     float* __restrict__ buf, int N) {
    const int wave = threadIdx.x >> 5, lane = threadIdx.x & 31;
    const int row = blockIdx.x * 8 + wave;
    if (row >= N) return;

    float4 xv = ((const float4*)(x + row * DIM))[lane];
    float4 mv = ((float4*)(buf + row * DIM))[lane];
    float4 bv = ((const float4*)bias)[lane];

    // mobius_matvec tail
    float xn = fmaxf(sqrtf(wave_sum(dot4(xv, xv))), MINN);
    float mn = fmaxf(sqrtf(wave_sum(dot4(mv, mv))), MINN);
    float mabs = wave_sum(fabsf(mv.x) + fabsf(mv.y) + fabsf(mv.z) + fabsf(mv.w));
    float s = tanhf(mn / xn * artanh_f(xn)) / mn;
    if (mabs == 0.0f) s = 0.0f;
    // proj(res): analytic norm = |s| * mn
    float rn = fmaxf(fabsf(s) * mn, MINN);
    if (rn > MAXN) s *= MAXN / rn;
    float4 res = scale4(mv, s);

    // hyp_bias = proj(expmap0(bias))
    float bn = fmaxf(sqrtf(wave_sum(dot4(bv, bv))), MINN);
    float bs = tanhf(bn) / bn;
    float hbn = fmaxf(tanhf(bn), MINN);
    if (hbn > MAXN) bs *= MAXN / hbn;
    float4 hb = scale4(bv, bs);

    // h = proj(mobius_add(res, hb))
    float X2 = wave_sum(dot4(res, res));
    float Y2 = wave_sum(dot4(hb, hb));
    float XY = wave_sum(dot4(res, hb));
    float ca = 1.0f + 2.0f * XY + Y2;
    float cb = 1.0f - X2;
    float den = fmaxf(1.0f + 2.0f * XY + X2 * Y2, MINN);
    float4 h = fma4(res, ca / den, hb, cb / den);
    float hn = fmaxf(sqrtf(wave_sum(dot4(h, h))), MINN);
    float hs = 1.0f;
    if (hn > MAXN) { hs = MAXN / hn; hn = MAXN; }
    // xt = logmap0(h)
    float ls = artanh_f(hn) / hn;
    float4 xt = scale4(h, hs * ls);
    ((float4*)(buf + row * DIM))[lane] = xt;
}

// ---------------------------------------------------------------- K3: degrees
__global__ __launch_bounds__(256) void k_deg_init(float* __restrict__ deg, int N) {
    int i = blockIdx.x * 256 + threadIdx.x;
    if (i < N) deg[i] = 1.0f;                 // self loop
}
__global__ __launch_bounds__(256) void k_deg_count(const int64_t* __restrict__ rows,
                                                   float* __restrict__ deg, int E) {
    int e = blockIdx.x * 256 + threadIdx.x;
    if (e < E)
        __hip_atomic_fetch_add(&deg[(int)rows[e]], 1.0f,
                               __ATOMIC_RELAXED, __HIP_MEMORY_SCOPE_AGENT);
}
__global__ __launch_bounds__(256) void k_deg_to_dis(float* __restrict__ deg, int N) {
    int i = blockIdx.x * 256 + threadIdx.x;
    if (i < N) deg[i] = rsqrtf(deg[i]);       // deg >= 1 always
}

// ---------------------------------------------------------------- K4: aggregation
__global__ __launch_bounds__(256) void k_agg_self(const float* __restrict__ xt,
                                                  const float* __restrict__ dis,
                                                  float* __restrict__ agg, int N) {
    int i = blockIdx.x * 256 + threadIdx.x;   // over N*DIM elements
    if (i < N * DIM) {
        float d = dis[i >> 7];
        agg[i] = d * d * xt[i];
    }
}
__global__ __launch_bounds__(256) void k_agg_edges(const int64_t* __restrict__ rows,
                                                   const int64_t* __restrict__ cols,
                                                   const float* __restrict__ dis,
                                                   const float* __restrict__ xt,
                                                   float* __restrict__ agg, int E) {
    const int e = blockIdx.x * 8 + (threadIdx.x >> 5);
    if (e >= E) return;
    const int lane = threadIdx.x & 31;
    const int r = (int)rows[e], c = (int)cols[e];
    const float s = dis[r] * dis[c];
    float4 v = ((const float4*)(xt + c * DIM))[lane];
    float* dst = agg + r * DIM + lane * 4;
    __hip_atomic_fetch_add(dst + 0, s * v.x, __ATOMIC_RELAXED, __HIP_MEMORY_SCOPE_AGENT);
    __hip_atomic_fetch_add(dst + 1, s * v.y, __ATOMIC_RELAXED, __HIP_MEMORY_SCOPE_AGENT);
    __hip_atomic_fetch_add(dst + 2, s * v.z, __ATOMIC_RELAXED, __HIP_MEMORY_SCOPE_AGENT);
    __hip_atomic_fetch_add(dst + 3, s * v.w, __ATOMIC_RELAXED, __HIP_MEMORY_SCOPE_AGENT);
}

// ---------------------------------------------------------------- K5: HypAct (in-place on d_out)
__global__ __launch_bounds__(256) void k_hyp_act(float* __restrict__ buf, int N) {
    const int wave = threadIdx.x >> 5, lane = threadIdx.x & 31;
    const int row = blockIdx.x * 8 + wave;
    if (row >= N) return;

    float4 a = ((float4*)(buf + row * DIM))[lane];
    // h = proj(expmap0(agg)); t = logmap0(h)   (track uniform scale analytically)
    float an = fmaxf(sqrtf(wave_sum(dot4(a, a))), MINN);
    float s1 = tanhf(an) / an;
    float ne = fmaxf(an * s1, MINN);
    if (ne > MAXN) { s1 *= MAXN / ne; ne = MAXN; }
    float t_s = s1 * artanh_f(ne) / ne;
    float4 t = scale4(a, t_s);
    // leaky relu (breaks uniform scaling -> need fresh norm)
    t.x = t.x >= 0.f ? t.x : 0.01f * t.x;
    t.y = t.y >= 0.f ? t.y : 0.01f * t.y;
    t.z = t.z >= 0.f ? t.z : 0.01f * t.z;
    t.w = t.w >= 0.f ? t.w : 0.01f * t.w;
    // out = proj(expmap0(t, c_out))
    float tn = fmaxf(sqrtf(wave_sum(dot4(t, t))), MINN);
    float s2 = tanhf(tn) / tn;
    float no = fmaxf(tn * s2, MINN);
    if (no > MAXN) s2 *= MAXN / no;
    ((float4*)(buf + row * DIM))[lane] = scale4(t, s2);
}

// ---------------------------------------------------------------- launch
extern "C" void kernel_launch(void* const* d_in, const int* in_sizes, int n_in,
                              void* d_out, int out_size, void* d_ws, size_t ws_size,
                              hipStream_t stream) {
    const float*   x    = (const float*)d_in[0];
    const float*   w    = (const float*)d_in[1];
    const float*   bias = (const float*)d_in[2];
    const int64_t* ei   = (const int64_t*)d_in[3];

    const int N = in_sizes[0] / DIM;
    const int E = in_sizes[3] / 2;
    const int64_t* rows = ei;
    const int64_t* cols = ei + E;

    // workspace: [ mx/xt : N*128 f32 ][ deg/dis : N f32 ][ wh : 128*128 f16 ]
    float*    xt  = (float*)d_ws;
    float*    deg = xt + (size_t)N * DIM;
    _Float16* wh  = (_Float16*)(deg + N);
    float*    agg = (float*)d_out;          // d_out doubles as agg buffer

    k_convert_w<<<(DIM * DIM + 255) / 256, 256, 0, stream>>>(w, wh, DIM * DIM);
    k_gemm_wmma<<<(N + 127) / 128, 256, 0, stream>>>(x, wh, xt, N);
    k_linear_tail<<<(N + 7) / 8, 256, 0, stream>>>(x, bias, xt, N);
    k_deg_init<<<(N + 255) / 256, 256, 0, stream>>>(deg, N);
    k_deg_count<<<(E + 255) / 256, 256, 0, stream>>>(rows, deg, E);
    k_deg_to_dis<<<(N + 255) / 256, 256, 0, stream>>>(deg, N);
    k_agg_self<<<((size_t)N * DIM + 255) / 256, 256, 0, stream>>>(xt, deg, agg, N);
    k_agg_edges<<<(E + 7) / 8, 256, 0, stream>>>(rows, cols, deg, xt, agg, E);
    k_hyp_act<<<(N + 7) / 8, 256, 0, stream>>>(agg, N);
}
